// Pointnet_sa_module_msg_85134841741753
// MI455X (gfx1250) — matrix operand order repaired and checked by reading, compiled, not run on hardware
//
#include <hip/hip_runtime.h>

#define B_      4
#define N_      16384
#define C_      64
#define NPOINT_ 1024

typedef __attribute__((ext_vector_type(16))) _Float16 v16h;
typedef __attribute__((ext_vector_type(8)))  _Float16 h8;
typedef __attribute__((ext_vector_type(8)))  float    v8f;
typedef __attribute__((ext_vector_type(4)))  int      v4i_;

// ---------------------------------------------------------------------------
// gfx1250 async global->LDS copy path (guarded; falls back to b128 load/store
// pairs if the builtin is absent). Builtin signature (from clang diagnostic):
//   void (int4 addrspace(1)*, int4 addrspace(3)*, imm offset, imm cpol)
#if defined(__has_builtin)
#if __has_builtin(__builtin_amdgcn_global_load_async_to_lds_b128) && \
    __has_builtin(__builtin_amdgcn_s_wait_asynccnt)
#define HAVE_ASYNC_LDS 1
#endif
#endif

struct b16blk { unsigned int x[4]; };

__device__ inline void copy16_g2l(void* lds_dst, const void* gsrc) {
#ifdef HAVE_ASYNC_LDS
  __builtin_amdgcn_global_load_async_to_lds_b128(
      (__attribute__((address_space(1))) v4i_*)(gsrc),
      (__attribute__((address_space(3))) v4i_*)(lds_dst), 0, 0);
#else
  *(b16blk*)lds_dst = *(const b16blk*)gsrc;
#endif
}

__device__ inline void async_copy_wait() {
#ifdef HAVE_ASYNC_LDS
  __builtin_amdgcn_s_wait_asynccnt(0);
#endif
}

// ---------------------------------------------------------------------------
// WMMA fragment loaders (CDNA5 wave32 layouts, CDNA5 ISA §7.12.2).
// A (16x32 f16, row-major LDS): lane m = lane&15, K-half = lane>>4.
// Two aligned 16B runs -> 2x ds_load_b128.
__device__ inline v16h load_frag_a(const _Float16* buf, int ld, int mbase, int kbase, int lane) {
  int m  = mbase + (lane & 15);
  int kh = (lane >> 4) * 8;
  const _Float16* p = buf + m * ld + kbase + kh;
  h8 lo = *(const h8*)(p);
  h8 hi = *(const h8*)(p + 16);
  return __builtin_shufflevector(lo, hi, 0, 1, 2, 3, 4, 5, 6, 7,
                                 8, 9, 10, 11, 12, 13, 14, 15);
}

// B fragments are pre-swizzled in global memory: block (kc,nt) holds 32 lanes
// x 16 contiguous halves, so a lane reads 32 contiguous bytes (2x b128, coalesced).
__device__ inline v16h load_frag_b_swz(const _Float16* gw, int NT, int kc, int nt, int lane) {
  const _Float16* p = gw + (((kc * NT + nt) * 32 + lane) << 4);
  return *(const v16h*)p;
}

// ---------------------------------------------------------------------------
// Weight pre-swizzle: w[cin][cout] f32 -> fragment-order f16 (zero-padded K).
// Element i: block = i>>9 (32 lanes x 16 halves), lane = (i>>4)&31, e = i&15,
// k = kc*32 + (lane>>4)*8 + (e&7) + (e>>3)*16, n = nt*16 + (lane&15).
__global__ __launch_bounds__(128) void swizzle_w_kernel(const float* __restrict__ w,
                                                        _Float16* __restrict__ dst,
                                                        int cin, int cinPad, int cout) {
  int i = blockIdx.x * 128 + threadIdx.x;
  int total = cinPad * cout;
  if (i >= total) return;
  int NT = cout >> 4;
  int block = i >> 9, within = i & 511;
  int lane = within >> 4, e = within & 15;
  int kc = block / NT, nt = block % NT;
  int k = kc * 32 + ((lane >> 4) << 3) + (e & 7) + ((e >> 3) << 4);
  int n = nt * 16 + (lane & 15);
  dst[i] = (k < cin) ? (_Float16)w[k * cout + n] : (_Float16)0.f;
}

// ---------------------------------------------------------------------------
// Kernel 1: sequential farthest-point sampling. One workgroup per batch,
// per-thread residual distances in registers, LDS argmax tree per step.
__global__ __launch_bounds__(1024) void fps_kernel(const float* __restrict__ xyz,
                                                   float* __restrict__ new_xyz,
                                                   int* __restrict__ fps_idx) {
  const int b   = blockIdx.x;
  const int tid = threadIdx.x;
  __shared__ float redv[1024];
  __shared__ int   redi[1024];
  __shared__ float sel[3];
  const float* xb = xyz + (size_t)b * N_ * 3;
  float dist[N_ / 1024];
#pragma unroll
  for (int p = 0; p < N_ / 1024; ++p) dist[p] = 1e10f;
  int far = 0;
  for (int it = 0; it < NPOINT_; ++it) {
    if (tid == 0) {
      fps_idx[b * NPOINT_ + it] = far;
      float x = xb[far * 3 + 0], y = xb[far * 3 + 1], z = xb[far * 3 + 2];
      new_xyz[(size_t)(b * NPOINT_ + it) * 3 + 0] = x;
      new_xyz[(size_t)(b * NPOINT_ + it) * 3 + 1] = y;
      new_xyz[(size_t)(b * NPOINT_ + it) * 3 + 2] = z;
      sel[0] = x; sel[1] = y; sel[2] = z;
    }
    __syncthreads();
    float sx = sel[0], sy = sel[1], sz = sel[2];
    float best = -1.0f; int bi = 0;
#pragma unroll
    for (int p = 0; p < N_ / 1024; ++p) {
      int g = p * 1024 + tid;
      float dx = xb[g * 3 + 0] - sx, dy = xb[g * 3 + 1] - sy, dz = xb[g * 3 + 2] - sz;
      float d2 = dx * dx + dy * dy + dz * dz;
      float nd = dist[p] < d2 ? dist[p] : d2;
      dist[p] = nd;
      if (nd > best) { best = nd; bi = g; }
    }
    redv[tid] = best; redi[tid] = bi;
    __syncthreads();
    for (int s = 512; s > 0; s >>= 1) {
      if (tid < s) {
        float ov = redv[tid + s]; int oi = redi[tid + s];
        if (ov > redv[tid] || (ov == redv[tid] && oi < redi[tid])) { redv[tid] = ov; redi[tid] = oi; }
      }
      __syncthreads();
    }
    far = redi[0];
  }
}

// ---------------------------------------------------------------------------
// Kernel 2: ball query. One wave32 per center; ballot-based ordered compaction
// preserves original index order; pad short groups with first-found index.
__global__ __launch_bounds__(128) void ball_query_kernel(const float* __restrict__ xyz,
                                                         const float* __restrict__ new_xyz,
                                                         int* __restrict__ idx,
                                                         int ns, float r2) {
  int center = blockIdx.x * 4 + (threadIdx.x >> 5);
  int lane   = threadIdx.x & 31;
  if (center >= B_ * NPOINT_) return;
  int b = center / NPOINT_;
  const float* xb = xyz + (size_t)b * N_ * 3;
  float cx = new_xyz[center * 3 + 0], cy = new_xyz[center * 3 + 1], cz = new_xyz[center * 3 + 2];
  int* out = idx + (size_t)center * ns;
  int cnt = 0, first = 0;
  for (int base = 0; base < N_; base += 32) {
    int g = base + lane;
    float dx = xb[g * 3 + 0] - cx, dy = xb[g * 3 + 1] - cy, dz = xb[g * 3 + 2] - cz;
    bool in = (dx * dx + dy * dy + dz * dz) < r2;
    unsigned mask = __builtin_amdgcn_ballot_w32(in);
    if (cnt == 0 && mask) first = base + __builtin_ctz(mask);
    if (in) {
      int pos = cnt + __popc(mask & ((1u << lane) - 1u));
      if (pos < ns) out[pos] = g;
    }
    cnt += __popc(mask);
    if (cnt >= ns) break;
  }
  for (int j = cnt + lane; j < ns; j += 32) out[j] = first;
}

// ---------------------------------------------------------------------------
// MLP layer: f16 activations in LDS (A frags = 2x ds_load_b128), pre-swizzled
// f16 weights streamed from global (B frags = 2x global_load_b128, L2-hot),
// f32 WMMA accumulate, bias+ReLU, f16 store. 4 waves split the output tiles.
template <int NS>
__device__ inline void mlp_layer(const _Float16* actIn, _Float16* actOut,
                                 const _Float16* __restrict__ gw,
                                 const float* __restrict__ bias,
                                 int cinPad, int cout, int tid) {
  int wave = tid >> 5, lane = tid & 31;
  const int MT = NS / 16, NT = cout / 16, KT = cinPad / 32;
  for (int t = wave; t < MT * NT; t += 4) {
    int mt = t / NT, nt = t % NT;
    v8f acc = {};
    for (int kc = 0; kc < KT; ++kc) {
      v16h a  = load_frag_a(actIn, 128, mt * 16, kc * 32, lane);
      v16h bf = load_frag_b_swz(gw, NT, kc, nt, lane);
      acc = __builtin_amdgcn_wmma_f32_16x16x32_f16(false, a, false, bf, (short)0, acc, false, false);
    }
    int n  = nt * 16 + (lane & 15);
    int mb = mt * 16 + (lane >> 4) * 8;
    float bv = bias[n];
#pragma unroll
    for (int j = 0; j < 8; ++j) {
      float v = acc[j] + bv;
      v = v > 0.f ? v : 0.f;
      actOut[(mb + j) * 128 + n] = (_Float16)v;
    }
  }
  __syncthreads();
}

// Kernel 3: fused gather + 3-layer MLP + max-pool. One workgroup per center.
template <int NS, int C1, int C2>
__global__ __launch_bounds__(128) void group_mlp_kernel(
    const float* __restrict__ xyz, const float* __restrict__ points,
    const float* __restrict__ new_xyz, const int* __restrict__ idx,
    const _Float16* __restrict__ gw0, const float* __restrict__ b0,
    const _Float16* __restrict__ gw1, const float* __restrict__ b1,
    const _Float16* __restrict__ gw2, const float* __restrict__ b2,
    _Float16* __restrict__ out) {
  constexpr int CIN = 3 + C_;  // 67
  constexpr int K0P = 96;      // layer-0 K padded to 3 chunks of 32
  constexpr int LD  = 128;
  __shared__ _Float16 actA[NS * LD];
  __shared__ _Float16 actB[NS * LD];
  __shared__ unsigned pool[128];
  __shared__ int      lidx[NS];
  const int tid    = threadIdx.x;
  const int center = blockIdx.x;
  const int b      = center / NPOINT_;
  const float* xb  = xyz + (size_t)b * N_ * 3;
  const float* pb  = points + (size_t)b * N_ * C_;
  float cx = new_xyz[center * 3 + 0], cy = new_xyz[center * 3 + 1], cz = new_xyz[center * 3 + 2];
  pool[tid] = 0u;  // blockDim == 128
  // Stage this center's neighbor indices into LDS (async path when available).
  if (tid < NS / 4) copy16_g2l(lidx + tid * 4, idx + (size_t)center * NS + tid * 4);
  async_copy_wait();
  __syncthreads();
  // Gather centered coords + point features into f16 LDS tile [NS][96]
  for (int e = tid; e < NS * K0P; e += 128) {
    int r = e / K0P, c = e % K0P;
    int id = lidx[r];
    float v;
    if (c < 3)        v = xb[id * 3 + c] - (c == 0 ? cx : (c == 1 ? cy : cz));
    else if (c < CIN) v = pb[id * C_ + (c - 3)];
    else              v = 0.f;
    actA[r * LD + c] = (_Float16)v;
  }
  __syncthreads();
  mlp_layer<NS>(actA, actB, gw0, b0, K0P, C1, tid);
  mlp_layer<NS>(actB, actA, gw1, b1, C1, C2, tid);
  // Final layer fused with max-pool straight out of the accumulators.
  {
    const int cout = 128;
    int wave = tid >> 5, lane = tid & 31;
    const int MT = NS / 16, NT = cout / 16, KT = C2 / 32;
    for (int t = wave; t < MT * NT; t += 4) {
      int mt = t / NT, nt = t % NT;
      v8f acc = {};
      for (int kc = 0; kc < KT; ++kc) {
        v16h a  = load_frag_a(actA, LD, mt * 16, kc * 32, lane);
        v16h bf = load_frag_b_swz(gw2, NT, kc, nt, lane);
        acc = __builtin_amdgcn_wmma_f32_16x16x32_f16(false, a, false, bf, (short)0, acc, false, false);
      }
      int n = nt * 16 + (lane & 15);
      float bv = b2[n];
      float mx = 0.f;
#pragma unroll
      for (int j = 0; j < 8; ++j) {
        float v = acc[j] + bv;
        v = v > 0.f ? v : 0.f;
        mx = mx > v ? mx : v;
      }
      // post-ReLU values >= 0 -> uint bit pattern is order-preserving
      atomicMax(&pool[n], __float_as_uint(mx));
    }
    __syncthreads();
    out[(size_t)center * 128 + tid] = (_Float16)__uint_as_float(pool[tid]);
  }
}

// ---------------------------------------------------------------------------
// Kernel 4: aggregation GEMM [4096,256] x [256,128] + bias + ReLU.
// Activations (already f16) async-copied to LDS; pre-swizzled weights streamed
// from global per fragment. 64-row tile per workgroup, 8 accumulators/wave.
__global__ __launch_bounds__(128) void agg_kernel(const _Float16* __restrict__ o0,
                                                  const _Float16* __restrict__ o1,
                                                  const _Float16* __restrict__ waggs,
                                                  const float* __restrict__ bagg,
                                                  float* __restrict__ newpts) {
  __shared__ _Float16 actL[64 * 256];
  const int tid  = threadIdx.x;
  const int row0 = blockIdx.x * 64;
  // 64 rows x 256 halves: row = concat(o0[row], o1[row]); 32 x 16B segs per row.
  for (int t = tid; t < 64 * 32; t += 128) {
    int r = t >> 5, seg = t & 31;
    int grow = row0 + r;
    const _Float16* src = (seg < 16) ? (o0 + (size_t)grow * 128 + seg * 8)
                                     : (o1 + (size_t)grow * 128 + (seg - 16) * 8);
    copy16_g2l(actL + r * 256 + seg * 8, src);
  }
  async_copy_wait();
  __syncthreads();
  const int wave = tid >> 5, lane = tid & 31;
  v8f zero = {};
  v8f acc[8];
#pragma unroll
  for (int i = 0; i < 8; ++i) acc[i] = zero;
  for (int kc = 0; kc < 8; ++kc) {
#pragma unroll
    for (int i = 0; i < 8; ++i) {
      int t = wave + i * 4;
      int mt = t >> 3, nt = t & 7;
      v16h a  = load_frag_a(actL, 256, mt * 16, kc * 32, lane);
      v16h bf = load_frag_b_swz(waggs, 8, kc, nt, lane);
      acc[i] = __builtin_amdgcn_wmma_f32_16x16x32_f16(false, a, false, bf, (short)0, acc[i], false, false);
    }
  }
#pragma unroll
  for (int i = 0; i < 8; ++i) {
    int t = wave + i * 4;
    int mt = t >> 3, nt = t & 7;
    int n  = nt * 16 + (lane & 15);
    int mb = mt * 16 + (lane >> 4) * 8;
    float bv = bagg[n];
#pragma unroll
    for (int j = 0; j < 8; ++j) {
      float v = acc[i][j] + bv;
      v = v > 0.f ? v : 0.f;
      newpts[(size_t)(row0 + mb + j) * 128 + n] = v;
    }
  }
}

// ---------------------------------------------------------------------------
extern "C" void kernel_launch(void* const* d_in, const int* in_sizes, int n_in,
                              void* d_out, int out_size, void* d_ws, size_t ws_size,
                              hipStream_t stream) {
  (void)in_sizes; (void)n_in; (void)out_size; (void)ws_size;
  const float* xyz    = (const float*)d_in[0];
  const float* points = (const float*)d_in[1];
  const float* w00 = (const float*)d_in[2];  const float* b00 = (const float*)d_in[3];
  const float* w01 = (const float*)d_in[4];  const float* b01 = (const float*)d_in[5];
  const float* w02 = (const float*)d_in[6];  const float* b02 = (const float*)d_in[7];
  const float* w10 = (const float*)d_in[8];  const float* b10 = (const float*)d_in[9];
  const float* w11 = (const float*)d_in[10]; const float* b11 = (const float*)d_in[11];
  const float* w12 = (const float*)d_in[12]; const float* b12 = (const float*)d_in[13];
  const float* wagg = (const float*)d_in[14]; const float* bagg = (const float*)d_in[15];

  float* out     = (float*)d_out;
  float* new_xyz = out;                                             // [4,1024,3]
  float* new_pts = out + (size_t)B_ * NPOINT_ * 3;                  // [4,1024,128]
  int*   fps_idx = (int*)(out + (size_t)B_ * NPOINT_ * (3 + 128));  // [4,1024]

  char* ws = (char*)d_ws;
  int*      idx0  = (int*)(ws);                  // 4096*32 ints   (512 KB)
  int*      idx1  = (int*)(ws + 524288);         // 4096*64 ints   (1 MB)
  _Float16* o0    = (_Float16*)(ws + 1572864);   // 4096*128 f16   (1 MB)
  _Float16* o1    = (_Float16*)(ws + 2621440);   // 4096*128 f16   (1 MB)
  _Float16* w00s  = (_Float16*)(ws + 3670016);   // 96x64
  _Float16* w01s  = (_Float16*)(ws + 3682304);   // 64x64
  _Float16* w02s  = (_Float16*)(ws + 3690496);   // 64x128
  _Float16* w10s  = (_Float16*)(ws + 3706880);   // 96x64
  _Float16* w11s  = (_Float16*)(ws + 3719168);   // 64x96
  _Float16* w12s  = (_Float16*)(ws + 3731456);   // 96x128
  _Float16* waggs = (_Float16*)(ws + 3756032);   // 256x128

  // Weight pre-swizzle into WMMA fragment order (runs while FPS serializes).
  swizzle_w_kernel<<<(96 * 64 + 127) / 128, 128, 0, stream>>>(w00, w00s, 67, 96, 64);
  swizzle_w_kernel<<<(64 * 64 + 127) / 128, 128, 0, stream>>>(w01, w01s, 64, 64, 64);
  swizzle_w_kernel<<<(64 * 128 + 127) / 128, 128, 0, stream>>>(w02, w02s, 64, 64, 128);
  swizzle_w_kernel<<<(96 * 64 + 127) / 128, 128, 0, stream>>>(w10, w10s, 67, 96, 64);
  swizzle_w_kernel<<<(64 * 96 + 127) / 128, 128, 0, stream>>>(w11, w11s, 64, 64, 96);
  swizzle_w_kernel<<<(96 * 128 + 127) / 128, 128, 0, stream>>>(w12, w12s, 96, 96, 128);
  swizzle_w_kernel<<<(256 * 128 + 127) / 128, 128, 0, stream>>>(wagg, waggs, 256, 256, 128);

  fps_kernel<<<B_, 1024, 0, stream>>>(xyz, new_xyz, fps_idx);
  ball_query_kernel<<<(B_ * NPOINT_) / 4, 128, 0, stream>>>(xyz, new_xyz, idx0, 32, 0.1f * 0.1f);
  ball_query_kernel<<<(B_ * NPOINT_) / 4, 128, 0, stream>>>(xyz, new_xyz, idx1, 64, 0.2f * 0.2f);
  group_mlp_kernel<32, 64, 64><<<B_ * NPOINT_, 128, 0, stream>>>(
      xyz, points, new_xyz, idx0, w00s, b00, w01s, b01, w02s, b02, o0);
  group_mlp_kernel<64, 64, 96><<<B_ * NPOINT_, 128, 0, stream>>>(
      xyz, points, new_xyz, idx1, w10s, b10, w11s, b11, w12s, b12, o1);
  agg_kernel<<<(B_ * NPOINT_) / 64, 128, 0, stream>>>(o0, o1, waggs, bagg, new_pts);
}
